// NeuralODE_68831145885972
// MI455X (gfx1250) — matrix operand order repaired and checked by reading, not compiled, not run on hardware
//
#include <hip/hip_runtime.h>
#include <math.h>

// ---------------------------------------------------------------------------
// NeuralODE (GRU encoder + VAE head + dopri5 over 4 intervals) for gfx1250.
// One persistent workgroup per 16 batch rows; all GEMMs via
// v_wmma_f32_16x16x32_f16 (f16 inputs, f32 accumulate).
// Round 4: weight pointers carry an opaque loop-variant offset (zoff==0 at
// runtime) so LICM/CSE can neither hoist them into scratch spills (round 2)
// nor did we need volatile (round 3: SCOPE_SYS flat loads + serializing
// waits). Loads stay plain global_load_b128, L2-resident, clause-batchable.
// ---------------------------------------------------------------------------

typedef __attribute__((ext_vector_type(16))) _Float16 v16h;
typedef __attribute__((ext_vector_type(8)))  float    v8f;
typedef __attribute__((ext_vector_type(4)))  unsigned int u32x4;

union V16H { v16h v; u32x4 q[2]; };

#define HDIM 256
#define LSTR 264   // padded f16 LDS row stride (halfs): 528B -> full bank spread

// ws layout (halfs)
#define OFF_WHH 0
#define OFF_EW1 196608
#define OFF_EW2 262144
#define OFF_OW1 393216
#define WS_HALFS 655360

__device__ __forceinline__ v8f vzero8() {
  v8f z = {0.f,0.f,0.f,0.f,0.f,0.f,0.f,0.f};
  return z;
}

__device__ __forceinline__ v8f wmma_f16(v16h a, v16h b, v8f c) {
  return __builtin_amdgcn_wmma_f32_16x16x32_f16(false, a, false, b, (short)0, c, false, false);
}

__device__ __forceinline__ float selu_f(float x) {
  const float al = 1.6732632423543772f, sc = 1.0507009873554805f;
  return sc * (x > 0.f ? x : al * (expf(x) - 1.f));
}

// GEMM: out tiles (16 rows x 16 cols each) = act(16xK=256, f16 LDS) @ W.T.
// W is (N x 256) row-major f16 in global. acc[i] is the 16x16 f32 D fragment
// for output-column tile tiles[i]. kt-outer: one A fragment live.
template<int NT>
__device__ __forceinline__ void gemm_tiles(const _Float16* __restrict__ act,
                                           const _Float16* __restrict__ Wg,
                                           const int* tiles, v8f* acc,
                                           int lm, int lh) {
  const int s = lh * 8;
#pragma unroll
  for (int kt = 0; kt < 8; ++kt) {
    V16H A;
    const u32x4* p = (const u32x4*)(act + lm * LSTR + kt * 32 + s);
    A.q[0] = p[0];   // K = kt*32 + s      .. +7
    A.q[1] = p[2];   // K = kt*32 + 16 + s .. +7
#pragma unroll
    for (int i = 0; i < NT; ++i) {
      V16H Bf;
      const u32x4* q =
          (const u32x4*)(Wg + (tiles[i] * 16 + lm) * HDIM + lh * 8 + kt * 32);
      Bf.q[0] = q[0];  // K = kt*32 + 8*lh       .. +7
      Bf.q[1] = q[2];  // K = kt*32 + 16 + 8*lh  .. +7
      acc[i] = wmma_f16(A.v, Bf.v, acc[i]);
    }
  }
}

// 4-layer ODE MLP: selu,selu,selu,linear. Input in bufA (f16 LDS), scratch
// ping-pong bufA<->bufB, result k (f32, D-layout) in kout for the wave's
// two feature tiles. Starts with a barrier so caller's bufA writes are seen.
// OWbase = OW1 (+ opaque zero); W2..W4 are at fixed 65536-half strides.
__device__ __forceinline__ void mlp4(_Float16* bufA, _Float16* bufB,
                                     const _Float16* __restrict__ OWbase,
                                     const float (*ob)[2], const int* tiles2,
                                     int lm, int lh, v8f* kout) {
  v8f acc[2];
  __syncthreads();
  acc[0] = vzero8(); acc[1] = vzero8();
  gemm_tiles<2>(bufA, OWbase, tiles2, acc, lm, lh);
#pragma unroll
  for (int i = 0; i < 2; ++i) { int col = tiles2[i] * 16 + lm;
#pragma unroll
    for (int r = 0; r < 8; ++r) { int m = r + 8 * lh;
      bufB[m * LSTR + col] = (_Float16)selu_f(acc[i][r] + ob[0][i]); } }
  __syncthreads();
  acc[0] = vzero8(); acc[1] = vzero8();
  gemm_tiles<2>(bufB, OWbase + 65536, tiles2, acc, lm, lh);
#pragma unroll
  for (int i = 0; i < 2; ++i) { int col = tiles2[i] * 16 + lm;
#pragma unroll
    for (int r = 0; r < 8; ++r) { int m = r + 8 * lh;
      bufA[m * LSTR + col] = (_Float16)selu_f(acc[i][r] + ob[1][i]); } }
  __syncthreads();
  acc[0] = vzero8(); acc[1] = vzero8();
  gemm_tiles<2>(bufA, OWbase + 131072, tiles2, acc, lm, lh);
#pragma unroll
  for (int i = 0; i < 2; ++i) { int col = tiles2[i] * 16 + lm;
#pragma unroll
    for (int r = 0; r < 8; ++r) { int m = r + 8 * lh;
      bufB[m * LSTR + col] = (_Float16)selu_f(acc[i][r] + ob[2][i]); } }
  __syncthreads();
  kout[0] = vzero8(); kout[1] = vzero8();
  gemm_tiles<2>(bufB, OWbase + 196608, tiles2, kout, lm, lh);
#pragma unroll
  for (int i = 0; i < 2; ++i) kout[i] = kout[i] + ob[3][i];
}

// ---------------------------------------------------------------------------
__global__ __launch_bounds__(256) void node_prep(
    const float* __restrict__ Whh, const float* __restrict__ eW1,
    const float* __restrict__ eW2, const float* __restrict__ o1,
    const float* __restrict__ o2, const float* __restrict__ o3,
    const float* __restrict__ o4, _Float16* __restrict__ ws) {
  int i = blockIdx.x * blockDim.x + threadIdx.x;
  if (i < 196608)        ws[i] = (_Float16)Whh[i];
  else if (i < 262144)   ws[i] = (_Float16)eW1[i - 196608];
  else if (i < 393216)   ws[i] = (_Float16)eW2[i - 262144];
  else if (i < 458752)   ws[i] = (_Float16)o1[i - 393216];
  else if (i < 524288)   ws[i] = (_Float16)o2[i - 458752];
  else if (i < 589824)   ws[i] = (_Float16)o3[i - 524288];
  else if (i < WS_HALFS) ws[i] = (_Float16)o4[i - 589824];
}

__global__ __launch_bounds__(256) void node_main(
    const float* __restrict__ x, const float* __restrict__ meta,
    const float* __restrict__ eps, const float* __restrict__ times,
    const float* __restrict__ doses, const float* __restrict__ Wih,
    const float* __restrict__ bih, const float* __restrict__ bhh,
    const float* __restrict__ eb1p, const float* __restrict__ eb2p,
    const float* __restrict__ ob1, const float* __restrict__ ob2,
    const float* __restrict__ ob3, const float* __restrict__ ob4,
    const float* __restrict__ fcW, const float* __restrict__ fcb,
    const _Float16* __restrict__ ws16, float* __restrict__ out,
    int zoff /* always 0 at runtime; opaque to the compiler */) {

  __shared__ float fbufA[16 * 256];                   // h (GRU) then y (ODE), f32
  __shared__ __align__(16) _Float16 g16A[16 * LSTR];  // f16 act ping
  __shared__ __align__(16) _Float16 g16B[16 * LSTR];  // f16 act pong
  __shared__ float kb[6][16 * 256];                   // k1..k6, per-lane private
  __shared__ float xmS[16][8];
  __shared__ float rowT[16], rowT1[16], rowDT[16], rowDTC[16], rowSum[16], rowDose[16];
  __shared__ int rowFlag[16], rowAcc[16];

  const int tid = threadIdx.x;
  const int w = tid >> 5, lane = tid & 31, lm = lane & 15, lh = lane >> 4;
  const int r0 = blockIdx.x * 16;

  const _Float16* Whh16 = ws16 + OFF_WHH;
  const _Float16* EW1 = ws16 + OFF_EW1;
  const _Float16* EW2 = ws16 + OFF_EW2;
  const _Float16* OW1 = ws16 + OFF_OW1;

  // ---- per-lane constant preloads ----
  int tilesG[6], tiles2[2], tiles4[4];
  float bih6[6], bhh6[6], wih6[6][8];
#pragma unroll
  for (int i = 0; i < 6; ++i) {
    tilesG[i] = w + 8 * i;                 // i = gate*2 + f2
    int n = tilesG[i] * 16 + lm;
    bih6[i] = bih[n]; bhh6[i] = bhh[n];
#pragma unroll
    for (int k = 0; k < 8; ++k) wih6[i][k] = Wih[n * 8 + k];
  }
  tiles2[0] = w; tiles2[1] = w + 8;
  tiles4[0] = w; tiles4[1] = w + 8; tiles4[2] = w + 16; tiles4[3] = w + 24;
  float e1b[2]; e1b[0] = eb1p[w * 16 + lm]; e1b[1] = eb1p[(w + 8) * 16 + lm];
  float e2b[4];
#pragma unroll
  for (int i = 0; i < 4; ++i) e2b[i] = eb2p[tiles4[i] * 16 + lm];
  float ob[4][2];
#pragma unroll
  for (int i = 0; i < 2; ++i) {
    int n = tiles2[i] * 16 + lm;
    ob[0][i] = ob1[n]; ob[1][i] = ob2[n]; ob[2][i] = ob3[n]; ob[3][i] = ob4[n];
  }

  // ---- init: h = 0, first xm tile ----
  for (int i = tid; i < 16 * 256; i += 256) fbufA[i] = 0.f;
  for (int i = tid; i < 16 * LSTR; i += 256) { g16A[i] = (_Float16)0.f; g16B[i] = (_Float16)0.f; }
  if (tid < 128) {
    int m = tid >> 3, k = tid & 7;
    xmS[m][k] = (k < 4) ? x[(r0 + m) * 256 + k] : meta[(r0 + m) * 4 + (k - 4)];
  }
  __syncthreads();

  // ======================= GRU (64 steps) =======================
  for (int t = 0; t < 64; ++t) {
    v8f acc6[6];
#pragma unroll
    for (int i = 0; i < 6; ++i) acc6[i] = vzero8();
    // opaque loop-variant base: defeats LICM weight caching (zoff==0)
    gemm_tiles<6>(g16A, Whh16 + zoff * (t + 1), tilesG, acc6, lm, lh);

    float hnew[2][8];
#pragma unroll
    for (int f2 = 0; f2 < 2; ++f2) {
      int col = (w + 8 * f2) * 16 + lm;
#pragma unroll
      for (int r = 0; r < 8; ++r) {
        int m = r + 8 * lh;
        float gr = bih6[0 + f2], gz = bih6[2 + f2], gn = bih6[4 + f2];
#pragma unroll
        for (int k = 0; k < 8; ++k) {
          float xv = xmS[m][k];
          gr += xv * wih6[0 + f2][k];
          gz += xv * wih6[2 + f2][k];
          gn += xv * wih6[4 + f2][k];
        }
        float hr = acc6[0 + f2][r] + bhh6[0 + f2];
        float hz = acc6[2 + f2][r] + bhh6[2 + f2];
        float hn = acc6[4 + f2][r] + bhh6[4 + f2];
        float rg = 1.f / (1.f + expf(-(gr + hr)));
        float zg = 1.f / (1.f + expf(-(gz + hz)));
        float ng = tanhf(gn + rg * hn);
        float hold = fbufA[m * 256 + col];
        hnew[f2][r] = (1.f - zg) * ng + zg * hold;
      }
    }
    __syncthreads();
#pragma unroll
    for (int f2 = 0; f2 < 2; ++f2) {
      int col = (w + 8 * f2) * 16 + lm;
#pragma unroll
      for (int r = 0; r < 8; ++r) {
        int m = r + 8 * lh;
        fbufA[m * 256 + col] = hnew[f2][r];
        g16A[m * LSTR + col] = (_Float16)hnew[f2][r];
      }
    }
    if (tid < 128 && t < 63) {
      int m = tid >> 3, k = tid & 7;
      xmS[m][k] = (k < 4) ? x[(r0 + m) * 256 + (t + 1) * 4 + k]
                          : meta[(r0 + m) * 4 + (k - 4)];
    }
    __syncthreads();
  }

  // ======================= encoder =======================
  {
    v8f a2[2]; a2[0] = vzero8(); a2[1] = vzero8();
    gemm_tiles<2>(g16A, EW1, tiles2, a2, lm, lh);
#pragma unroll
    for (int i = 0; i < 2; ++i) { int col = tiles2[i] * 16 + lm;
#pragma unroll
      for (int r = 0; r < 8; ++r) { int m = r + 8 * lh;
        float v = a2[i][r] + e1b[i]; v = v > 0.f ? v : 0.f;
        g16B[m * LSTR + col] = (_Float16)v; } }
    __syncthreads();
    v8f a4[4];
#pragma unroll
    for (int i = 0; i < 4; ++i) a4[i] = vzero8();
    gemm_tiles<4>(g16B, EW2, tiles4, a4, lm, lh);
#pragma unroll
    for (int f2 = 0; f2 < 2; ++f2) { int col = (w + 8 * f2) * 16 + lm;
#pragma unroll
      for (int r = 0; r < 8; ++r) { int m = r + 8 * lh;
        float mean = a4[f2][r] + e2b[f2];
        float stdv = a4[2 + f2][r] + e2b[2 + f2];
        float ev = eps[(r0 + m) * 256 + col];
        fbufA[m * 256 + col] = ev * stdv + mean; } }
    __syncthreads();
  }

  // ======================= dopri5 x 4 intervals =======================
  const float cA21 = 0.2f;
  const float cA31 = (float)(3.0/40.0), cA32 = (float)(9.0/40.0);
  const float cA41 = (float)(44.0/45.0), cA42 = (float)(-56.0/15.0), cA43 = (float)(32.0/9.0);
  const float cA51 = (float)(19372.0/6561.0), cA52 = (float)(-25360.0/2187.0),
              cA53 = (float)(64448.0/6561.0), cA54 = (float)(-212.0/729.0);
  const float cA61 = (float)(9017.0/3168.0), cA62 = (float)(-355.0/33.0),
              cA63 = (float)(46732.0/5247.0), cA64 = (float)(49.0/176.0),
              cA65 = (float)(-5103.0/18656.0);
  const float B0 = (float)(35.0/384.0), B2 = (float)(500.0/1113.0),
              B3 = (float)(125.0/192.0), B4c = (float)(-2187.0/6784.0),
              B5 = (float)(11.0/84.0);
  const float E0 = (float)(35.0/384.0 - 5179.0/57600.0),
              E2 = (float)(500.0/1113.0 - 7571.0/16695.0),
              E3 = (float)(125.0/192.0 - 393.0/640.0),
              E4 = (float)(-2187.0/6784.0 + 92097.0/339200.0),
              E5 = (float)(11.0/84.0 - 187.0/2100.0),
              E6 = (float)(-1.0/40.0);

  // store a D-layout register pair into a per-lane-private f32 LDS k-buffer
  auto kstore = [&](float* kbuf, const v8f* kk) {
#pragma unroll
    for (int f2 = 0; f2 < 2; ++f2) { int col = (w + 8 * f2) * 16 + lm;
#pragma unroll
      for (int r = 0; r < 8; ++r) { int m = r + 8 * lh;
        kbuf[m * 256 + col] = kk[f2][r]; } }
  };

  for (int itv = 0; itv < 4; ++itv) {
    if (tid < 16) {
      int b = r0 + tid;
      float t0 = times[(itv * 1024 + b) * 2 + 0];
      float t1v = times[(itv * 1024 + b) * 2 + 1];
      rowT[tid] = t0; rowT1[tid] = t1v;
      rowDT[tid] = fmaxf(t1v - t0, 1e-6f) * 0.1f;
      rowDose[tid] = (t0 < t1v) ? doses[itv * 1024 + b] : 0.f;
    }
    __syncthreads();
    for (int i = tid; i < 16 * 256; i += 256) fbufA[i] += rowDose[i >> 8];
    __syncthreads();

    for (int st = 0; st < 32; ++st) {
      if (tid < 16) {
        float tt = rowT[tid], t1v = rowT1[tid];
        rowFlag[tid] = (tt < t1v) ? 1 : 0;
        rowDTC[tid] = fminf(fmaxf(rowDT[tid], 0.f), fmaxf(t1v - tt, 0.f));
        rowSum[tid] = 0.f;
      }
      __syncthreads();
      int any = 0;
#pragma unroll
      for (int m = 0; m < 16; ++m) any |= rowFlag[m];
      if (!any) break;

      // distinct opaque offset per stage: defeats LICM over st AND CSE
      // across the 7 same-weight evaluations within one step (zoff==0)
      const _Float16* OWs = OW1 + zoff * (st * 8);

      v8f kreg[2], k7[2], y5v[2];

      // stage 1: input = y
      for (int i = tid; i < 16 * 256; i += 256)
        g16A[(i >> 8) * LSTR + (i & 255)] = (_Float16)fbufA[i];
      mlp4(g16A, g16B, OWs + zoff * 1, ob, tiles2, lm, lh, kreg);
      kstore(kb[0], kreg);

      // stage 2
#pragma unroll
      for (int f2 = 0; f2 < 2; ++f2) { int col = (w + 8 * f2) * 16 + lm;
#pragma unroll
        for (int r = 0; r < 8; ++r) { int m = r + 8 * lh; int e = m * 256 + col;
          float d = rowDTC[m];
          float vv = fbufA[e] + d * (cA21 * kb[0][e]);
          g16A[m * LSTR + col] = (_Float16)vv; } }
      mlp4(g16A, g16B, OWs + zoff * 2, ob, tiles2, lm, lh, kreg);
      kstore(kb[1], kreg);

      // stage 3
#pragma unroll
      for (int f2 = 0; f2 < 2; ++f2) { int col = (w + 8 * f2) * 16 + lm;
#pragma unroll
        for (int r = 0; r < 8; ++r) { int m = r + 8 * lh; int e = m * 256 + col;
          float d = rowDTC[m];
          float vv = fbufA[e] + d * (cA31 * kb[0][e] + cA32 * kb[1][e]);
          g16A[m * LSTR + col] = (_Float16)vv; } }
      mlp4(g16A, g16B, OWs + zoff * 3, ob, tiles2, lm, lh, kreg);
      kstore(kb[2], kreg);

      // stage 4
#pragma unroll
      for (int f2 = 0; f2 < 2; ++f2) { int col = (w + 8 * f2) * 16 + lm;
#pragma unroll
        for (int r = 0; r < 8; ++r) { int m = r + 8 * lh; int e = m * 256 + col;
          float d = rowDTC[m];
          float vv = fbufA[e] +
                     d * (cA41 * kb[0][e] + cA42 * kb[1][e] + cA43 * kb[2][e]);
          g16A[m * LSTR + col] = (_Float16)vv; } }
      mlp4(g16A, g16B, OWs + zoff * 4, ob, tiles2, lm, lh, kreg);
      kstore(kb[3], kreg);

      // stage 5
#pragma unroll
      for (int f2 = 0; f2 < 2; ++f2) { int col = (w + 8 * f2) * 16 + lm;
#pragma unroll
        for (int r = 0; r < 8; ++r) { int m = r + 8 * lh; int e = m * 256 + col;
          float d = rowDTC[m];
          float vv = fbufA[e] +
                     d * (cA51 * kb[0][e] + cA52 * kb[1][e] +
                          cA53 * kb[2][e] + cA54 * kb[3][e]);
          g16A[m * LSTR + col] = (_Float16)vv; } }
      mlp4(g16A, g16B, OWs + zoff * 5, ob, tiles2, lm, lh, kreg);
      kstore(kb[4], kreg);

      // stage 6
#pragma unroll
      for (int f2 = 0; f2 < 2; ++f2) { int col = (w + 8 * f2) * 16 + lm;
#pragma unroll
        for (int r = 0; r < 8; ++r) { int m = r + 8 * lh; int e = m * 256 + col;
          float d = rowDTC[m];
          float vv = fbufA[e] +
                     d * (cA61 * kb[0][e] + cA62 * kb[1][e] + cA63 * kb[2][e] +
                          cA64 * kb[3][e] + cA65 * kb[4][e]);
          g16A[m * LSTR + col] = (_Float16)vv; } }
      mlp4(g16A, g16B, OWs + zoff * 6, ob, tiles2, lm, lh, kreg);
      kstore(kb[5], kreg);

      // y5 and stage 7
#pragma unroll
      for (int f2 = 0; f2 < 2; ++f2) { int col = (w + 8 * f2) * 16 + lm;
#pragma unroll
        for (int r = 0; r < 8; ++r) { int m = r + 8 * lh; int e = m * 256 + col;
          float d = rowDTC[m];
          float y5 = fbufA[e] +
                     d * (B0 * kb[0][e] + B2 * kb[2][e] + B3 * kb[3][e] +
                          B4c * kb[4][e] + B5 * kb[5][e]);
          y5v[f2][r] = y5;
          g16A[m * LSTR + col] = (_Float16)y5; } }
      mlp4(g16A, g16B, OWs + zoff * 7, ob, tiles2, lm, lh, k7);

      // error norm
      float part[8];
#pragma unroll
      for (int r = 0; r < 8; ++r) part[r] = 0.f;
#pragma unroll
      for (int f2 = 0; f2 < 2; ++f2) { int col = (w + 8 * f2) * 16 + lm;
#pragma unroll
        for (int r = 0; r < 8; ++r) { int m = r + 8 * lh; int e = m * 256 + col;
          float d = rowDTC[m];
          float yv = fbufA[e];
          float err = d * (E0 * kb[0][e] + E2 * kb[2][e] + E3 * kb[3][e] +
                           E4 * kb[4][e] + E5 * kb[5][e] + E6 * k7[f2][r]);
          float sc = 1e-6f + 1e-3f * fmaxf(fabsf(yv), fabsf(y5v[f2][r]));
          float q = err / sc;
          part[r] += q * q; } }
#pragma unroll
      for (int r = 0; r < 8; ++r) {
        float v = part[r];
        v += __shfl_xor(v, 1, 32);
        v += __shfl_xor(v, 2, 32);
        v += __shfl_xor(v, 4, 32);
        v += __shfl_xor(v, 8, 32);
        if (lm == 0) atomicAdd(&rowSum[r + 8 * lh], v);
      }
      __syncthreads();
      if (tid < 16) {
        float en = sqrtf(rowSum[tid] * (1.f / 256.f));
        int act = rowFlag[tid];
        int acc = (en <= 1.f) && act;
        rowAcc[tid] = acc;
        if (acc) rowT[tid] += rowDTC[tid];
        float fac = 0.9f * powf(en + 1e-10f, -0.2f);
        fac = fminf(fmaxf(fac, 0.2f), 10.f);
        if (act) rowDT[tid] = fmaxf(rowDTC[tid], 1e-8f) * fac;
      }
      __syncthreads();
#pragma unroll
      for (int f2 = 0; f2 < 2; ++f2) { int col = (w + 8 * f2) * 16 + lm;
#pragma unroll
        for (int r = 0; r < 8; ++r) { int m = r + 8 * lh;
          if (rowAcc[m]) fbufA[m * 256 + col] = y5v[f2][r]; } }
      __syncthreads();
    }
  }

  // ======================= head =======================
  __syncthreads();
  if (tid < 16) {
    float acc = fcb[0];
    for (int j = 0; j < 256; ++j) acc += fbufA[tid * 256 + j] * fcW[j];
    for (int j = 0; j < 4; ++j) acc += meta[(r0 + tid) * 4 + j] * fcW[256 + j];
    out[r0 + tid] = acc;
  }
}

extern "C" void kernel_launch(void* const* d_in, const int* in_sizes, int n_in,
                              void* d_out, int out_size, void* d_ws, size_t ws_size,
                              hipStream_t stream) {
  (void)in_sizes; (void)n_in; (void)out_size; (void)ws_size;
  const float* x     = (const float*)d_in[0];
  const float* meta  = (const float*)d_in[1];
  const float* eps   = (const float*)d_in[2];
  const float* times = (const float*)d_in[3];
  const float* doses = (const float*)d_in[4];
  const float* gWih  = (const float*)d_in[5];
  const float* gWhh  = (const float*)d_in[6];
  const float* gbih  = (const float*)d_in[7];
  const float* gbhh  = (const float*)d_in[8];
  const float* eW1   = (const float*)d_in[9];
  const float* eb1   = (const float*)d_in[10];
  const float* eW2   = (const float*)d_in[11];
  const float* eb2   = (const float*)d_in[12];
  const float* oW1   = (const float*)d_in[13];
  const float* ob1   = (const float*)d_in[14];
  const float* oW2   = (const float*)d_in[15];
  const float* ob2   = (const float*)d_in[16];
  const float* oW3   = (const float*)d_in[17];
  const float* ob3   = (const float*)d_in[18];
  const float* oW4   = (const float*)d_in[19];
  const float* ob4   = (const float*)d_in[20];
  const float* fcW   = (const float*)d_in[21];
  const float* fcb   = (const float*)d_in[22];
  _Float16* ws16 = (_Float16*)d_ws;

  node_prep<<<WS_HALFS / 256, 256, 0, stream>>>(gWhh, eW1, eW2, oW1, oW2, oW3, oW4, ws16);
  node_main<<<64, 256, 0, stream>>>(x, meta, eps, times, doses,
                                    gWih, gbih, gbhh, eb1, eb2,
                                    ob1, ob2, ob3, ob4, fcW, fcb,
                                    ws16, (float*)d_out, /*zoff=*/0);
}
